// FactorizedSpectralConv_57793079935489
// MI455X (gfx1250) — matrix operand order repaired and checked by reading, compiled, not run on hardware
//
#include <hip/hip_runtime.h>
#include <hip/hip_bf16.h>

// ---------------------------------------------------------------------------
// FNO FactorizedSpectralConv for MI455X (gfx1250, wave32, WMMA).
// B=16, C=64, H=W=128, modes 16x16 (two corners). Full FFT replaced by
// partial-DFT GEMMs through v_wmma_f32_16x16x32_f16. All WMMA operands that
// live in memory are stored in *fragment-linear* order: per 16x32/32x16 tile,
// element order = [lane][16 halves] matching the ISA VGPR layout, so each
// lane loads its fragment as one contiguous 32-byte access (b128 pairs).
// ---------------------------------------------------------------------------

typedef __attribute__((ext_vector_type(16))) _Float16 v16h;
typedef __attribute__((ext_vector_type(8)))  _Float16 v8h;
typedef __attribute__((ext_vector_type(8)))  float    v8f;
typedef __attribute__((ext_vector_type(16))) float    v16f;

#define WMMA16(a, b, c) \
  __builtin_amdgcn_wmma_f32_16x16x32_f16(false, (a), false, (b), (short)0, (c), false, false)

// ---- fragment-linear index math (CDNA5 ISA 7.12.2, wave32) -----------------
// A (16x32 f16): lane = hi*16+m with hi=(k>>3)&1; half q = k<16 ? k-hi*8
//                : 8+(k-16)-hi*8.  offset = lane*16+q.
__device__ __forceinline__ int a_frag_off(int m, int k) {
  const int hi = (k >> 3) & 1;
  const int q = (k < 16) ? (k - hi * 8) : (8 + (k - 16) - hi * 8);
  return ((hi * 16 + m) << 4) + q;
}
// B (32x16 f16): lane = ((k>>3)&1)*16+n; half q = ((k>>4)&1)*8+(k&7). offset = lane*16+q.
__device__ __forceinline__ int b_frag_off(int k, int n) {
  return (((((k >> 3) & 1) * 16 + n)) << 4) + (((k >> 4) & 1) * 8 + (k & 7));
}
// packed-order decode (for generator kernels): e in [0,512)
__device__ __forceinline__ void a_frag_decode(int e, int& m, int& k) {
  const int lane = e >> 4, q = e & 15, hi = lane >> 4;
  m = lane & 15;
  k = (q < 8) ? (hi * 8 + q) : (16 + hi * 8 + (q - 8));
}
__device__ __forceinline__ void b_frag_decode(int e, int& k, int& n) {
  const int lane = e >> 4, q = e & 15, hi = lane >> 4;
  n = lane & 15;
  k = (q < 8) ? (hi * 8 + q) : (16 + hi * 8 + (q - 8));
}

// ---- fragment loads/stores --------------------------------------------------
__device__ __forceinline__ v16h load_frag_packed(const _Float16* p) {
  return *(const v16h*)(p + ((threadIdx.x & 31) << 4));   // 32B/lane
}
__device__ __forceinline__ v16h load_frag_packed_f32(const float* p) {
  const v16f f = *(const v16f*)(p + ((threadIdx.x & 31) << 4));  // 64B/lane
  v16h h;
#pragma unroll
  for (int j = 0; j < 16; ++j) h[j] = (_Float16)f[j];
  return h;
}
__device__ __forceinline__ v16h cat8(v8h lo, v8h hi) {
  v16h r;
#pragma unroll
  for (int j = 0; j < 8; ++j) { r[j] = lo[j]; r[8 + j] = hi[j]; }
  return r;
}
// D fragment (16x16 f32): lane holds col n=lane&15, rows m=(lane>>4)*8+v.
template <typename F>
__device__ __forceinline__ void store_d_frag(const v8f& d, F&& put) {
  const int lane = threadIdx.x & 31;
  const int n = lane & 15, hi = lane >> 4;
#pragma unroll
  for (int v = 0; v < 8; ++v) put(hi * 8 + v, n, d[v]);
}
// store 8 consecutive halves (one D row-run) as a 16B vector
__device__ __forceinline__ void store_row8(const v8f& d, _Float16* base) {
  v8h h;
#pragma unroll
  for (int v = 0; v < 8; ++v) h[v] = (_Float16)d[v];
  *(v8h*)base = h;
}
// generic elementwise A loader (only for the raw-weight fallback path)
template <typename F>
__device__ __forceinline__ v16h load_a_frag(F&& get) {
  const int lane = threadIdx.x & 31;
  const int m = lane & 15, hi = lane >> 4;
  v16h a;
#pragma unroll
  for (int p = 0; p < 8; ++p) {
    const int k = (p < 4) ? (hi * 8 + 2 * p) : (16 + hi * 8 + 2 * (p - 4));
    a[2 * p]     = get(m, k);
    a[2 * p + 1] = get(m, k + 1);
  }
  return a;
}

// ---- workspace layout (bytes) ----------------------------------------------
static constexpr size_t OFF_FY = 0;                 // f16 packed [kt4][ri2][512]
static constexpr size_t OFF_FX = 8192;              // f16 packed [mt4][kt8][512]
static constexpr size_t OFF_GX = 40960;             // f16 packed [mt16][kt2][512]
static constexpr size_t OFF_GY = 73728;             // f16 packed [nt8][512]
static constexpr size_t OFF_XF = 81920;             // f16 packed [512 modes][kt4][512] (2 MB)
static constexpr size_t OFF_YW = OFF_XF + 2097152;  // f32 packed [1024 bc][ri2][512] (4 MB)
static constexpr size_t OFF_WT = OFF_YW + 4194304;  // f16 packed [512 modes][mt8*kt4][512] (16 MB)
static constexpr size_t WS_FULL = OFF_WT + 16777216;

// ---------------------------------------------------------------------------
// Twiddles, generated directly in fragment-linear order.
// Forward DFT e^{-i 2pi k x/128}; inverse e^{+i...} with hermitian doubling
// (x2 for ky>=1) folded into GY.
// ---------------------------------------------------------------------------
__global__ void fno_twiddles(_Float16* __restrict__ fy, _Float16* __restrict__ fx,
                             _Float16* __restrict__ gx, _Float16* __restrict__ gy) {
  const float K = 6.28318530717958647692f / 128.0f;
  const int tid = blockIdx.x * blockDim.x + threadIdx.x;
  const int nth = gridDim.x * blockDim.x;
  // FY (B-op of stage-A GEMM1): rows y, 2 N-tiles {cos, -sin}, 4 K-tiles
  for (int i = tid; i < 4096; i += nth) {
    const int kt = i >> 10, ri = (i >> 9) & 1, e = i & 511;
    int k, n; b_frag_decode(e, k, n);
    const int y = kt * 32 + k;
    const float a = K * (float)((y * n) & 127);
    fy[i] = (_Float16)(ri ? -__sinf(a) : __cosf(a));
  }
  // FX (A-op of stage-A GEMM2, 64x256): [[cos, sin],[-sin, cos]]
  for (int i = tid; i < 16384; i += nth) {
    const int f = i >> 9, mt = f >> 3, kt = f & 7, e = i & 511;
    int m, kl; a_frag_decode(e, m, kl);
    const int r = mt * 16 + m, c = kt * 32 + kl;
    const int rr = r & 31;
    const int kx = (rr < 16) ? rr : rr + 96;  // 0..15, 112..127
    const float a = K * (float)(((kx * (c & 127)) & 127));
    const float v = (r < 32) ? ((c < 128) ? __cosf(a) : __sinf(a))
                             : ((c < 128) ? -__sinf(a) : __cosf(a));
    fx[i] = (_Float16)v;
  }
  // GX (A-op of stage-C GEMM1, 256x64): [[cos, -sin],[sin, cos]]
  for (int i = tid; i < 16384; i += nth) {
    const int f = i >> 9, mt = f >> 1, kt = f & 1, e = i & 511;
    int m, kl; a_frag_decode(e, m, kl);
    const int r = mt * 16 + m, c = kt * 32 + kl;
    const int cc = c & 31;
    const int kx = (cc < 16) ? cc : cc + 96;
    const float a = K * (float)(((kx * (r & 127)) & 127));
    const float v = (r < 128) ? ((c < 32) ? __cosf(a) : -__sinf(a))
                              : ((c < 32) ? __sinf(a) : __cosf(a));
    gx[i] = (_Float16)v;
  }
  // GY (B-op of stage-C GEMM2, 32x128): rows<16 wt*cos, rows>=16 -wt*sin
  for (int i = tid; i < 4096; i += nth) {
    const int nt = i >> 9, e = i & 511;
    int k, n; b_frag_decode(e, k, n);
    const int ky = k & 15, y = nt * 16 + n;
    const float wt = ky ? 2.0f : 1.0f;
    const float a = K * (float)((ky * y) & 127);
    gy[i] = (_Float16)((k < 16) ? wt * __cosf(a) : -wt * __sinf(a));
  }
}

// ---------------------------------------------------------------------------
// Weight repack: (C,C,16,16) f32 x4 -> per-mode 128x128 f16 Wri blocks
// [[wr,-wi],[wi,wr]] in A-frag packed order. LDS-tiled so global reads stay
// coalesced. grid = 2 * 128(ct tiles) * 8(lm tiles), 256 threads (32x8).
// ---------------------------------------------------------------------------
__device__ __forceinline__ size_t wri_off(int mode, int row, int col) {
  return (size_t)mode * 16384 +
         (size_t)(((row >> 4) * 4 + (col >> 5)) * 512 + a_frag_off(row & 15, col & 31));
}

__global__ void fno_wtrans(const float* __restrict__ w0r, const float* __restrict__ w0i,
                           const float* __restrict__ w1r, const float* __restrict__ w1i,
                           _Float16* __restrict__ wt) {
  __shared__ float sr[32][33];
  __shared__ float si[32][33];
  const int blk = blockIdx.x;
  const int lmT = blk & 7;
  const int ctT = (blk >> 3) & 127;
  const int half = blk >> 10;
  const float* wr = half ? w1r : w0r;
  const float* wi = half ? w1i : w0i;
  const int tx = threadIdx.x & 31, ty = threadIdx.x >> 5;
  for (int r = ty; r < 32; r += 8) {
    const size_t off = (size_t)(ctT * 32 + r) * 256 + (lmT * 32 + tx);
    sr[r][tx] = wr[off];
    si[r][tx] = wi[off];
  }
  __syncthreads();
  for (int r = ty; r < 32; r += 8) {
    const int lm = lmT * 32 + r;        // local mode = kxl*16 + ky
    const int ct = ctT * 32 + tx;       // c*64 + t
    const int c = ct >> 6, t = ct & 63;
    const int mode = ((half ? 16 : 0) + (lm >> 4)) * 16 + (lm & 15);
    const float vr = sr[tx][r];
    const float vi = si[tx][r];
    wt[wri_off(mode, c, t)]            = (_Float16)vr;
    wt[wri_off(mode, 64 + c, 64 + t)]  = (_Float16)vr;
    wt[wri_off(mode, c, 64 + t)]       = (_Float16)(-vi);
    wt[wri_off(mode, 64 + c, t)]       = (_Float16)vi;
  }
}

// ---------------------------------------------------------------------------
// Stage A: partial forward DFT per (b,t).
// GEMM1: T1(128x32) = X(128x128) * FY ; GEMM2: Xf(64x16) = FX * [T1r;T1i].
// grid = 1024 (b*C + t), 256 threads (8 waves).
// ---------------------------------------------------------------------------
__launch_bounds__(256)
__global__ void fno_stage_a(const float* __restrict__ X, const _Float16* __restrict__ fyp,
                            const _Float16* __restrict__ fxp, _Float16* __restrict__ xf) {
  constexpr int XS = 136;                 // halves per row (16B aligned rows)
  __shared__ _Float16 xl[128 * XS];       // 34816 B
  __shared__ _Float16 t1p[8 * 512];       // 8192 B, B-frag packed [kt8][512]
  const int tid = threadIdx.x, wave = tid >> 5;
  const int lane = tid & 31, nn = lane & 15, hi = lane >> 4;
  const int b = blockIdx.x >> 6, t = blockIdx.x & 63;
  const float* Xp = X + (size_t)(b * 64 + t) * (128 * 128);
  for (int i = tid; i < 128 * 128; i += 256)
    xl[(i >> 7) * XS + (i & 127)] = (_Float16)Xp[i];
  __syncthreads();
  {  // GEMM1: wave = M tile; both complex halves of FY
    const int m0 = wave * 16;
    v8f accR = {}, accI = {};
    for (int kk = 0; kk < 4; ++kk) {
      const int k0 = kk * 32;
      const _Float16* src = &xl[(m0 + nn) * XS + k0 + hi * 8];
      const v16h a = cat8(*(const v8h*)src, *(const v8h*)(src + 16));
      const v16h bR = load_frag_packed(fyp + (kk * 2 + 0) * 512);
      const v16h bI = load_frag_packed(fyp + (kk * 2 + 1) * 512);
      accR = WMMA16(a, bR, accR);
      accI = WMMA16(a, bI, accI);
    }
    // T1 rows kg<128 = T1r[x], kg>=128 = T1i[x]; store D row-runs of 8 halves.
    const int kgR = wave * 16 + hi * 8;
    store_row8(accR, &t1p[(kgR >> 5) * 512 + ((((kgR >> 3) & 1) * 16 + nn) << 4) + (((kgR >> 4) & 1) * 8 + (kgR & 7))]);
    const int kgI = 128 + wave * 16 + hi * 8;
    store_row8(accI, &t1p[(kgI >> 5) * 512 + ((((kgI >> 3) & 1) * 16 + nn) << 4) + (((kgI >> 4) & 1) * 8 + (kgI & 7))]);
  }
  __syncthreads();
  if (wave < 4) {  // GEMM2: 4 M tiles of the 64x16 result
    const int m0 = wave * 16;
    v8f acc = {};
    for (int kk = 0; kk < 8; ++kk) {
      const v16h a = load_frag_packed(fxp + (wave * 8 + kk) * 512);
      const v16h bb = *(const v16h*)(&t1p[kk * 512 + (lane << 4)]);
      acc = WMMA16(a, bb, acc);
    }
    store_d_frag(acc, [&](int m, int n, float v) {
      const int row = m0 + m;                    // 0..63: [XfR(32); XfI(32)]
      const int reim = row >> 5, kx = row & 31;  // n = ky
      const int kg = reim * 64 + t;              // stage-B B-matrix row
      xf[(size_t)(kx * 16 + n) * 2048 + (kg >> 5) * 512 + b_frag_off(kg & 31, b)] =
          (_Float16)(v * 6.103515625e-05f);      // 1/(H*W), norm="forward"
    });
  }
}

// ---------------------------------------------------------------------------
// Stage B: per-mode channel mix. D(128x16) = Wri(128x128) * XfRI(128x16).
// grid = 512 modes, 8 waves = 8 M tiles, N = batch = 16 (one WMMA tile).
// ---------------------------------------------------------------------------
__launch_bounds__(256)
__global__ void fno_stage_b_wt(const _Float16* __restrict__ wt,
                               const _Float16* __restrict__ xf, float* __restrict__ yw) {
  const int wave = threadIdx.x >> 5;
  const int mode = blockIdx.x;
  const int kx = mode >> 4, ky = mode & 15;
  const _Float16* wb = wt + (size_t)mode * 16384;
  const _Float16* xb = xf + (size_t)mode * 2048;
  const int m0 = wave * 16;
  const int bfo = b_frag_off(kx, ky);  // block-uniform scatter offset into YW
  v8f acc = {};
  for (int kk = 0; kk < 4; ++kk) {
    const v16h a  = load_frag_packed(wb + (wave * 4 + kk) * 512);
    const v16h bb = load_frag_packed(xb + kk * 512);
    acc = WMMA16(a, bb, acc);
  }
  store_d_frag(acc, [&](int m, int n, float v) {
    const int mg = m0 + m, c = mg & 63, reim = mg >> 6;  // n = batch
    yw[(size_t)(n * 64 + c) * 1024 + reim * 512 + bfo] = v;
  });
}

// Fallback (small workspace): read f32 weights directly, elementwise.
__launch_bounds__(256)
__global__ void fno_stage_b_raw(const float* __restrict__ w0r, const float* __restrict__ w0i,
                                const float* __restrict__ w1r, const float* __restrict__ w1i,
                                const _Float16* __restrict__ xf, float* __restrict__ yw) {
  const int wave = threadIdx.x >> 5;
  const int mode = blockIdx.x;
  const int kx = mode >> 4, ky = mode & 15, kxl = kx & 15;
  const float* wr = (kx < 16) ? w0r : w1r;
  const float* wi = (kx < 16) ? w0i : w1i;
  const _Float16* xb = xf + (size_t)mode * 2048;
  const int m0 = wave * 16;
  const int bfo = b_frag_off(kx, ky);
  v8f acc = {};
  for (int kk = 0; kk < 4; ++kk) {
    const int k0 = kk * 32;
    const v16h a = load_a_frag([&](int m, int k) -> _Float16 {
      const int mg = m0 + m, kg = k0 + k;
      const int c = mg & 63, rowIm = mg >> 6;
      const int t = kg & 63, colIm = kg >> 6;
      const size_t off = ((size_t)(c * 64 + t) * 16 + kxl) * 16 + ky;
      const float v = rowIm ? (colIm ? wr[off] : wi[off]) : (colIm ? -wi[off] : wr[off]);
      return (_Float16)v;
    });
    const v16h bb = load_frag_packed(xb + kk * 512);
    acc = WMMA16(a, bb, acc);
  }
  store_d_frag(acc, [&](int m, int n, float v) {
    const int mg = m0 + m, c = mg & 63, reim = mg >> 6;
    yw[(size_t)(n * 64 + c) * 1024 + reim * 512 + bfo] = v;
  });
}

// ---------------------------------------------------------------------------
// Stage C: inverse partial DFT per (b,c).
// GEMM1: U(256x16) = GX(256x64) * [Yr;Yi](64x16)   (kx sum)
// GEMM2: out(128x128) = [Ur|Ui](128x32) * GY(32x128)  (ky sum + Re())
// grid = 1024 (b*C + c), 256 threads (8 waves).
// ---------------------------------------------------------------------------
__launch_bounds__(256)
__global__ void fno_stage_c(const float* __restrict__ yw, const _Float16* __restrict__ gxp,
                            const _Float16* __restrict__ gyp, float* __restrict__ out) {
  __shared__ _Float16 u[256 * 16];  // 8 KB, U row-major stride 16 (32B rows)
  const int tid = threadIdx.x, wave = tid >> 5;
  const int lane = tid & 31, nn = lane & 15, hi = lane >> 4;
  const int b = blockIdx.x >> 6, c = blockIdx.x & 63;
  const float* yb = yw + (size_t)(b * 64 + c) * 1024;
  for (int hlf = 0; hlf < 2; ++hlf) {  // 16 M tiles over 8 waves
    const int mt = wave + hlf * 8;
    const int m0 = mt * 16;
    v8f acc = {};
    for (int kt = 0; kt < 2; ++kt) {
      const v16h a  = load_frag_packed(gxp + (mt * 2 + kt) * 512);
      const v16h bb = load_frag_packed_f32(yb + kt * 512);
      acc = WMMA16(a, bb, acc);
    }
    store_d_frag(acc, [&](int m, int n, float v) { u[((m0 + m) << 4) + n] = (_Float16)v; });
  }
  __syncthreads();
  const int m0 = wave * 16;
  // A fragment of [Ur | Ui] (K=32): two aligned 16B LDS runs per lane.
  const v16h a = cat8(*(const v8h*)&u[((m0 + nn) << 4) + hi * 8],
                      *(const v8h*)&u[((128 + m0 + nn) << 4) + hi * 8]);
  float* ob = out + (size_t)(b * 64 + c) * (128 * 128);
  for (int nt = 0; nt < 8; ++nt) {
    const v16h bb = load_frag_packed(gyp + nt * 512);
    v8f acc = {};
    acc = WMMA16(a, bb, acc);
    store_d_frag(acc, [&](int m, int n, float v) { ob[(m0 + m) * 128 + nt * 16 + n] = v; });
  }
}

// ---------------------------------------------------------------------------
extern "C" void kernel_launch(void* const* d_in, const int* in_sizes, int n_in,
                              void* d_out, int out_size, void* d_ws, size_t ws_size,
                              hipStream_t stream) {
  (void)in_sizes; (void)n_in; (void)out_size;
  const float* X   = (const float*)d_in[0];
  const float* w0r = (const float*)d_in[1];
  const float* w0i = (const float*)d_in[2];
  const float* w1r = (const float*)d_in[3];
  const float* w1i = (const float*)d_in[4];
  float* out = (float*)d_out;
  char* ws = (char*)d_ws;

  _Float16* FY = (_Float16*)(ws + OFF_FY);
  _Float16* FX = (_Float16*)(ws + OFF_FX);
  _Float16* GX = (_Float16*)(ws + OFF_GX);
  _Float16* GY = (_Float16*)(ws + OFF_GY);
  _Float16* XF = (_Float16*)(ws + OFF_XF);
  float*    YW = (float*)(ws + OFF_YW);
  _Float16* WT = (_Float16*)(ws + OFF_WT);

  fno_twiddles<<<40, 256, 0, stream>>>(FY, FX, GX, GY);
  fno_stage_a<<<16 * 64, 256, 0, stream>>>(X, FY, FX, XF);
  if (ws_size >= WS_FULL) {
    fno_wtrans<<<2048, 256, 0, stream>>>(w0r, w0i, w1r, w1i, WT);
    fno_stage_b_wt<<<512, 256, 0, stream>>>(WT, XF, YW);
  } else {
    fno_stage_b_raw<<<512, 256, 0, stream>>>(w0r, w0i, w1r, w1i, XF, YW);
  }
  fno_stage_c<<<16 * 64, 256, 0, stream>>>(YW, GX, GY, out);
}
